// HGNNPConv_GIB_v2_90546500534481
// MI455X (gfx1250) — compile-verified, matrix operand order checked by reading, not run on hardware
//
#include <hip/hip_runtime.h>
#include <hip/hip_bf16.h>
#include <math.h>

#define NV   16384
#define NE   4096
#define CC   256
#define NNZK 131072
#define LDA  264   // padded LDS row stride in halves (256 + 8): avoids bank conflicts

typedef __attribute__((ext_vector_type(16))) _Float16 v16h;
typedef __attribute__((ext_vector_type(8)))  float    v8f;

union FragU { v16h h; float4 q[2]; };

// ---------------------------------------------------------------------------
// WMMA fragment loaders (CDNA5 ISA 7.12.2, 16-bit A 16x32 layout, wave32)
//   lanes 0-15 : row = row0+lane,    halves 0-7 = K k0..k0+7,  8-15 = k0+16..+23
//   lanes 16-31: row = row0+lane-16, halves 0-7 = K k0+8..+15, 8-15 = k0+24..+31
// ---------------------------------------------------------------------------
__device__ __forceinline__ v16h load_frag_g(const _Float16* __restrict__ base,
                                            int row0, int k0, int ld) {
  int lane = threadIdx.x & 31;
  int r  = row0 + (lane & 15);
  int kb = k0 + ((lane >> 4) << 3);
  const _Float16* p = base + (long)r * ld + kb;
  FragU u;
  u.q[0] = *reinterpret_cast<const float4*>(p);
  u.q[1] = *reinterpret_cast<const float4*>(p + 16);
  return u.h;
}

// fragment from LDS-staged, padded tile (emits ds_load_b128)
__device__ __forceinline__ v16h load_frag_s(const _Float16* s, int row0, int k0) {
  int lane = threadIdx.x & 31;
  int r  = row0 + (lane & 15);
  int kb = k0 + ((lane >> 4) << 3);
  const _Float16* p = s + r * LDA + kb;
  FragU u;
  u.q[0] = *reinterpret_cast<const float4*>(p);
  u.q[1] = *reinterpret_cast<const float4*>(p + 16);
  return u.h;
}

__device__ __forceinline__ v8f wmma_f16(v16h a, v16h b, v8f c) {
  // 8 args: (neg_a, A, neg_b, B, c_mod, C, reuse_a, reuse_b)
  return __builtin_amdgcn_wmma_f32_16x16x32_f16(false, a, false, b, (short)0, c,
                                                false, false);
}

// gfx1250 async copy: LDS[lds_off] <= MEM[gptr], 16 bytes, tracked by ASYNCcnt
__device__ __forceinline__ void async_copy16(const _Float16* gptr, unsigned lds_off) {
  asm volatile("global_load_async_to_lds_b128 %0, %1, off"
               :: "v"(lds_off), "v"(gptr) : "memory");
}

__device__ __forceinline__ void wait_async0() {
  asm volatile("s_wait_asynccnt 0" ::: "memory");
}

// ---------------------------------------------------------------------------
// small utility kernels
// ---------------------------------------------------------------------------
__global__ void zero_kernel(float* __restrict__ p, long n) {
  long t = (long)blockIdx.x * blockDim.x + threadIdx.x;
  if (t < n) p[t] = 0.0f;
}

__global__ void cvt_f16_kernel(const float* __restrict__ src,
                               _Float16* __restrict__ dst, long n) {
  long t = (long)blockIdx.x * blockDim.x + threadIdx.x;
  if (t < n) dst[t] = (_Float16)src[t];
}

__global__ void deg_kernel(const int* __restrict__ vi, const int* __restrict__ ei,
                           float* __restrict__ d_v, float* __restrict__ d_e, int nnz) {
  int i = blockIdx.x * blockDim.x + threadIdx.x;
  if (i < nnz) {
    atomicAdd(&d_v[vi[i]], 1.0f);
    atomicAdd(&d_e[ei[i]], 1.0f);
  }
}

// dst[dst_idx[i]] += src[src_idx[i]]  over 256-ch rows, 4 channels per thread
__global__ void scatter4_kernel(const float* __restrict__ src,
                                const int* __restrict__ src_idx,
                                const int* __restrict__ dst_idx,
                                float* __restrict__ dst, int nnz) {
  long t = (long)blockIdx.x * blockDim.x + threadIdx.x;
  if (t >= (long)nnz * 64) return;
  int i  = (int)(t >> 6);
  int c4 = (int)(t & 63) << 2;
  const float4 v = *reinterpret_cast<const float4*>(src + (long)src_idx[i] * CC + c4);
  float* d = dst + (long)dst_idx[i] * CC + c4;
  atomicAdd(d + 0, v.x);
  atomicAdd(d + 1, v.y);
  atomicAdd(d + 2, v.z);
  atomicAdd(d + 3, v.w);
}

// mat[r,c] /= max(deg[r], 1)
__global__ void rowdiv_kernel(float* __restrict__ mat, const float* __restrict__ deg,
                              long total) {
  long t = (long)blockIdx.x * blockDim.x + threadIdx.x;
  if (t < total) mat[t] /= fmaxf(deg[t >> 8], 1.0f);
}

// Xr = relu(Xv / max(d_v,1))
__global__ void xr_kernel(const float* __restrict__ Xv, const float* __restrict__ d_v,
                          float* __restrict__ Xr, long total) {
  long t = (long)blockIdx.x * blockDim.x + threadIdx.x;
  if (t < total) {
    float v = Xv[t] / fmaxf(d_v[t >> 8], 1.0f);
    Xr[t] = fmaxf(v, 0.0f);
  }
}

// per-row: s = row*att -> f16, norm = ||s||; one 256-thread block per row
__global__ __launch_bounds__(256) void rownorm_kernel(const float* __restrict__ rows,
                                                      const float* __restrict__ att,
                                                      _Float16* __restrict__ outh,
                                                      float* __restrict__ norms) {
  __shared__ float red[256];
  int r = blockIdx.x, c = threadIdx.x;
  float s = rows[(long)r * CC + c] * att[c];
  outh[(long)r * CC + c] = (_Float16)s;
  red[c] = s * s;
  __syncthreads();
  for (int off = 128; off > 0; off >>= 1) {
    if (c < off) red[c] += red[c + off];
    __syncthreads();
  }
  if (c == 0) norms[r] = sqrtf(red[0]);
}

// new_H[vi,ei] = 1 (alpha=0.5 => H=1 entries always pass threshold)
__global__ void seth_kernel(const int* __restrict__ vi, const int* __restrict__ ei,
                            float* __restrict__ newH, int nnz) {
  int i = blockIdx.x * blockDim.x + threadIdx.x;
  if (i < nnz) newH[(long)vi[i] * NE + ei[i]] = 1.0f;
}

// ---------------------------------------------------------------------------
// GEMM 1: Xt = X @ W^T + b   (f16 WMMA, f32 accumulate; small, direct-global)
// block = 8 waves -> 128x64 tile; wave -> 32x32 (2x2 fragments)
// ---------------------------------------------------------------------------
__global__ __launch_bounds__(256) void gemm_theta_kernel(
    const _Float16* __restrict__ Xh, const _Float16* __restrict__ Wh,
    const float* __restrict__ bias, float* __restrict__ Xt) {
  int w = threadIdx.x >> 5, lane = threadIdx.x & 31;
  int mbase = blockIdx.x * 128 + (w & 3) * 32;
  int cbase = blockIdx.y * 64 + (w >> 2) * 32;
  v8f acc[2][2] = {{{0}, {0}}, {{0}, {0}}};
  for (int k0 = 0; k0 < CC; k0 += 32) {
    v16h a0 = load_frag_g(Xh, mbase,      k0, CC);
    v16h a1 = load_frag_g(Xh, mbase + 16, k0, CC);
    v16h b0 = load_frag_g(Wh, cbase,      k0, CC);
    v16h b1 = load_frag_g(Wh, cbase + 16, k0, CC);
    acc[0][0] = wmma_f16(a0, b0, acc[0][0]);
    acc[0][1] = wmma_f16(a0, b1, acc[0][1]);
    acc[1][0] = wmma_f16(a1, b0, acc[1][0]);
    acc[1][1] = wmma_f16(a1, b1, acc[1][1]);
  }
  int mrow = (lane >> 4) << 3;   // +0 or +8
  int ncol = lane & 15;
  for (int i = 0; i < 2; ++i)
    for (int j = 0; j < 2; ++j) {
      int col = cbase + j * 16 + ncol;
      float bv = bias[col];
      for (int v = 0; v < 8; ++v) {
        int m = mbase + i * 16 + mrow + v;
        Xt[(long)m * CC + col] = acc[i][j][v] + bv;
      }
    }
}

// ---------------------------------------------------------------------------
// GEMM 2 (dominant): num = Xs @ Zs^T, LDS-staged via async-to-LDS copies.
// block = 8 waves -> 128(M) x 128(E) tile; 132KB LDS panels (full K=256).
// wave (4 in M x 2 in E) -> 32x64 tile: acc[2][4], 8 wmma per 6 frag loads.
// Fused epilogue:
//   A     = num / max(nx*nz, 1e-6)
//   new_H = (A > 0.8)            (H=1 entries fixed up by seth_kernel after)
//   loss += [ xlogy(A,2A) + xlogy(1-A,2(1-A)) ] / N
// ---------------------------------------------------------------------------
__global__ __launch_bounds__(256) void gemm_adj_kernel(
    const _Float16* __restrict__ Xs, const _Float16* __restrict__ Zs,
    const float* __restrict__ nx, const float* __restrict__ nz,
    float* __restrict__ newH, float* __restrict__ loss) {
  __shared__ float red[256];
  extern __shared__ _Float16 smem[];           // A: 128*LDA, B: 128*LDA halves
  _Float16* As = smem;
  _Float16* Bs = smem + 128 * LDA;

  int w = threadIdx.x >> 5, lane = threadIdx.x & 31;
  int mblk = blockIdx.x * 128;
  int eblk = blockIdx.y * 128;

  // cooperative async stage of both panels: 16B chunks, no VGPR round-trip
  for (int t = threadIdx.x; t < 128 * 32; t += 256) {
    int r = t >> 5;                 // panel row 0..127
    int c = (t & 31) << 3;          // k offset in halves (multiple of 8)
    async_copy16(Xs + (long)(mblk + r) * CC + c,
                 (unsigned)(size_t)(As + r * LDA + c));
    async_copy16(Zs + (long)(eblk + r) * CC + c,
                 (unsigned)(size_t)(Bs + r * LDA + c));
  }
  wait_async0();
  __syncthreads();

  int wm = (w & 3) * 32;            // wave M offset in block tile
  int we = (w >> 2) * 64;           // wave E offset in block tile

  v8f acc[2][4] = {{{0}, {0}, {0}, {0}}, {{0}, {0}, {0}, {0}}};
  for (int k0 = 0; k0 < CC; k0 += 32) {
    v16h a0 = load_frag_s(As, wm,      k0);
    v16h a1 = load_frag_s(As, wm + 16, k0);
    v16h b0 = load_frag_s(Bs, we,      k0);
    v16h b1 = load_frag_s(Bs, we + 16, k0);
    v16h b2 = load_frag_s(Bs, we + 32, k0);
    v16h b3 = load_frag_s(Bs, we + 48, k0);
    acc[0][0] = wmma_f16(a0, b0, acc[0][0]);
    acc[0][1] = wmma_f16(a0, b1, acc[0][1]);
    acc[0][2] = wmma_f16(a0, b2, acc[0][2]);
    acc[0][3] = wmma_f16(a0, b3, acc[0][3]);
    acc[1][0] = wmma_f16(a1, b0, acc[1][0]);
    acc[1][1] = wmma_f16(a1, b1, acc[1][1]);
    acc[1][2] = wmma_f16(a1, b2, acc[1][2]);
    acc[1][3] = wmma_f16(a1, b3, acc[1][3]);
  }

  int mrow = (lane >> 4) << 3;
  int ncol = lane & 15;
  float klsum = 0.0f;
  for (int i = 0; i < 2; ++i) {
    float nxv[8];
    for (int v = 0; v < 8; ++v) nxv[v] = nx[mblk + wm + i * 16 + mrow + v];
    for (int j = 0; j < 4; ++j) {
      int e = eblk + we + j * 16 + ncol;
      float nzv = nz[e];
      for (int v = 0; v < 8; ++v) {
        int m = mblk + wm + i * 16 + mrow + v;
        float denom = fmaxf(nxv[v] * nzv, 1e-6f);
        float A = acc[i][j][v] / denom;
        newH[(long)m * NE + e] = (A > 0.8f) ? 1.0f : 0.0f;
        float kl = 0.0f;
        if (A > 0.0f) kl += A * __logf(2.0f * A);
        float om = 1.0f - A;
        if (om > 0.0f) kl += om * __logf(2.0f * om);
        klsum += kl;
      }
    }
  }
  red[threadIdx.x] = klsum;
  __syncthreads();
  for (int off = 128; off > 0; off >>= 1) {
    if (threadIdx.x < off) red[threadIdx.x] += red[threadIdx.x + off];
    __syncthreads();
  }
  if (threadIdx.x == 0) atomicAdd(loss, red[0] * (1.0f / (float)NV));
}

// ---------------------------------------------------------------------------
extern "C" void kernel_launch(void* const* d_in, const int* in_sizes, int n_in,
                              void* d_out, int out_size, void* d_ws, size_t ws_size,
                              hipStream_t stream) {
  const float* X   = (const float*)d_in[0];
  const float* W   = (const float*)d_in[1];
  const float* b   = (const float*)d_in[2];
  const float* att = (const float*)d_in[3];
  const int*   vi  = (const int*)d_in[4];
  const int*   ei  = (const int*)d_in[5];

  float* out  = (float*)d_out;
  float* Xr   = out;                         // (N, C)
  float* loss = out + (size_t)NV * CC;       // scalar
  float* newH = loss + 1;                    // (N, E)

  // workspace carve-up (256B aligned)
  char* wsp = (char*)d_ws;
  size_t off = 0;
  auto carve = [&](size_t bytes) -> void* {
    void* p = wsp + off;
    off = (off + bytes + 255) & ~(size_t)255;
    return p;
  };
  _Float16* Xh  = (_Float16*)carve((size_t)NV * CC * 2);
  _Float16* Wh  = (_Float16*)carve((size_t)CC * CC * 2);
  float*    Xt  = (float*)   carve((size_t)NV * CC * 4);
  float*    Ye  = (float*)   carve((size_t)NE * CC * 4);
  float*    Xv  = (float*)   carve((size_t)NV * CC * 4);
  float*    Ze  = (float*)   carve((size_t)NE * CC * 4);
  _Float16* Xsh = (_Float16*)carve((size_t)NV * CC * 2);
  _Float16* Zsh = (_Float16*)carve((size_t)NE * CC * 2);
  float*    d_e = (float*)   carve((size_t)NE * 4);
  float*    d_v = (float*)   carve((size_t)NV * 4);
  float*    nx  = (float*)   carve((size_t)NV * 4);
  float*    nz  = (float*)   carve((size_t)NE * 4);

  auto gs = [](long n) { return dim3((unsigned)((n + 255) / 256)); };

  // zero accumulators
  zero_kernel<<<gs(NE), 256, 0, stream>>>(d_e, NE);
  zero_kernel<<<gs(NV), 256, 0, stream>>>(d_v, NV);
  zero_kernel<<<gs((long)NE * CC), 256, 0, stream>>>(Ye, (long)NE * CC);
  zero_kernel<<<gs((long)NV * CC), 256, 0, stream>>>(Xv, (long)NV * CC);
  zero_kernel<<<gs((long)NE * CC), 256, 0, stream>>>(Ze, (long)NE * CC);
  zero_kernel<<<gs(1), 256, 0, stream>>>(loss, 1);

  // f16 conversions for WMMA GEMMs
  cvt_f16_kernel<<<gs((long)NV * CC), 256, 0, stream>>>(X, Xh, (long)NV * CC);
  cvt_f16_kernel<<<gs((long)CC * CC), 256, 0, stream>>>(W, Wh, (long)CC * CC);

  // Xt = X @ W^T + b
  gemm_theta_kernel<<<dim3(NV / 128, CC / 64), 256, 0, stream>>>(Xh, Wh, b, Xt);

  // degrees
  deg_kernel<<<gs(NNZK), 256, 0, stream>>>(vi, ei, d_v, d_e, NNZK);

  // Ye = seg(Xt[vi], ei) / max(d_e,1)
  scatter4_kernel<<<gs((long)NNZK * 64), 256, 0, stream>>>(Xt, vi, ei, Ye, NNZK);
  rowdiv_kernel<<<gs((long)NE * CC), 256, 0, stream>>>(Ye, d_e, (long)NE * CC);

  // Xv = seg(Ye[ei], vi);  Xr = relu(Xv / max(d_v,1))
  scatter4_kernel<<<gs((long)NNZK * 64), 256, 0, stream>>>(Ye, ei, vi, Xv, NNZK);
  xr_kernel<<<gs((long)NV * CC), 256, 0, stream>>>(Xv, d_v, Xr, (long)NV * CC);

  // Ze = seg(Xr[vi], ei)
  scatter4_kernel<<<gs((long)NNZK * 64), 256, 0, stream>>>(Xr, vi, ei, Ze, NNZK);

  // Xs = Xr*att (f16) + nx;  Zs = Ze*att (f16) + nz
  rownorm_kernel<<<NV, 256, 0, stream>>>(Xr, att, Xsh, nx);
  rownorm_kernel<<<NE, 256, 0, stream>>>(Ze, att, Zsh, nz);

  // dense cosine-sim GEMM + fused threshold / KL loss (LDS-staged, async copies)
  size_t lds_bytes = (size_t)2 * 128 * LDA * sizeof(_Float16);   // 135168 B
  gemm_adj_kernel<<<dim3(NV / 128, NE / 128), 256, lds_bytes, stream>>>(
      Xsh, Zsh, nx, nz, newH, loss);

  // incidence entries of H always pass threshold at alpha=0.5
  seth_kernel<<<gs(NNZK), 256, 0, stream>>>(vi, ei, newH, NNZK);
}
